// Decoder_71880572665909
// MI455X (gfx1250) — matrix operand order repaired and checked by reading, not compiled, not run on hardware
//
#include <hip/hip_runtime.h>
#include <hip/hip_bf16.h>

// ---------------------------------------------------------------------------
// CDNA5 (gfx1250) decoder implementation.
// wave32, WMMA f32_16x16x32_f16 for all GEMMs, f32 accumulation.
// Async global->LDS staging (ASYNCcnt) for the f16 attention-score GEMMs.
// ---------------------------------------------------------------------------

typedef __attribute__((ext_vector_type(16))) _Float16 v16h;
typedef __attribute__((ext_vector_type(8)))  _Float16 v8h;
typedef __attribute__((ext_vector_type(4)))  _Float16 v4h;
typedef __attribute__((ext_vector_type(8)))  float    v8f;

#define LPITCH 40   // 32 halves payload + 8 halves pad; row stride 80B (16B aligned)
#define USE_ASYNC_COPY 1

union Frag { v16h v; v8h h8[2]; };

// A-fragment (16x32 f16, contiguous k order, matched by frag_b): lane lo = row, halves e:
//   k = e + hi*16  ->  two 16B chunks at +hi*16 and +hi*16+8
__device__ __forceinline__ v16h frag_a(const _Float16* tile, int rowTileBase, int lo, int hi) {
  Frag f;
  const _Float16* p = tile + (rowTileBase + lo) * LPITCH + hi * 16;
  f.h8[0] = *(const v8h*)(p);
  f.h8[1] = *(const v8h*)(p + 8);
  return f.v;
}
// B-fragment (32x16 f16): lane lo = col, k = e + hi*16 -> two 16B chunks
__device__ __forceinline__ v16h frag_b(const _Float16* tile, int colTileBase, int lo, int hi) {
  Frag f;
  const _Float16* p = tile + (colTileBase + lo) * LPITCH + hi * 16;
  f.h8[0] = *(const v8h*)(p);
  f.h8[1] = *(const v8h*)(p + 8);
  return f.v;
}

// 16-byte async copy global -> LDS (GLOBAL_LOAD_ASYNC_TO_LDS_B128, ASYNCcnt).
// Generic pointers to LDS carry the LDS byte offset in addr[31:0] (ISA 10.2).
__device__ __forceinline__ void async_cp16(const void* g, void* lds) {
#if USE_ASYNC_COPY
  unsigned off = (unsigned)(unsigned long long)lds;
  asm volatile("global_load_async_to_lds_b128 %0, %1, off"
               :: "v"(off), "v"(g) : "memory");
#else
  *(v8h*)lds = *(const v8h*)g;
#endif
}
__device__ __forceinline__ void async_wait() {
#if USE_ASYNC_COPY
  asm volatile("s_wait_asynccnt 0" ::: "memory");
#endif
}

// ---------------------------------------------------------------------------
// Generic 64x128-tile GEMM:  C = act( A @ op(B) + bias ) [+ residual]
//   BWT=true : B is (N,K) row-major, compute A @ B^T   (all weight matrices)
//   BWT=false: B is (K,N) row-major, compute A @ B     (P @ V)
//   ACT: 0 none | 1 leaky_relu(0.01) | 2 *scale | 3 *scale + causal mask(-1e7)
// 8 waves; each wave owns 2 M-tiles x 2 N-tiles (4 WMMA per 32-K chunk).
// blockIdx.z = batch index (strided pointers). K must be a multiple of 32.
// ---------------------------------------------------------------------------
struct GemmP {
  const void* A; long lda, bsA;
  const void* B; long ldb, bsB;
  const float* bias;
  const float* res; long ldres, bsRes;
  void* C; long ldc, bsC;
  int M, N, K;
  float scale;
};

template<typename TA, typename TB, typename TC, bool BWT, int ACT>
__global__ void __launch_bounds__(256) gemm_kernel(GemmP p) {
  __shared__ _Float16 As[64 * LPITCH];    // 64 rows  x 32 k
  __shared__ _Float16 Bs[128 * LPITCH];   // 128 cols x 32 k  (stored [n][k])

  const int tid  = threadIdx.x;
  const int lane = tid & 31;
  const int wave = tid >> 5;
  const int lo   = lane & 15;
  const int hi   = lane >> 4;
  const int z    = blockIdx.z;

  const TA* A = (const TA*)p.A + p.bsA * z;
  const TB* B = (const TB*)p.B + p.bsB * z;
  TC*       C = (TC*)p.C + p.bsC * z;

  const int rowBase = blockIdx.y * 64;
  const int colBase = blockIdx.x * 128;
  const int waveM = wave >> 2;     // 0..1 -> M-tile pair
  const int waveN = wave & 3;      // 0..3 -> N-tile pair

  v8f acc[2][2] = {};

  for (int k0 = 0; k0 < p.K; k0 += 32) {
    __syncthreads();

    // ---- stage A (64 x 32) ----
    if (sizeof(TA) == 2) {                 // f16 source: 16B async copies
#if !USE_ASYNC_COPY
#endif
      int r = tid >> 2, c8 = tid & 3;      // 256 chunks of 8 halves
      int gm = rowBase + r;
      _Float16* dst = &As[r * LPITCH + c8 * 8];
      if (gm < p.M) {
        async_cp16((const void*)((const _Float16*)A + (long)gm * p.lda + k0 + c8 * 8), dst);
      } else {
        *(v8h*)dst = (v8h)(_Float16)0.f;
      }
    } else {                               // f32 source: float4 load + v4h pack
#pragma unroll
      for (int it = 0; it < 2; ++it) {
        int i = tid + it * 256;            // 512 float4 chunks
        int r = i >> 3, c4 = i & 7;
        int gm = rowBase + r;
        float4 f = make_float4(0.f, 0.f, 0.f, 0.f);
        if (gm < p.M) f = *(const float4*)((const float*)A + (long)gm * p.lda + k0 + c4 * 4);
        v4h hh = {(_Float16)f.x, (_Float16)f.y, (_Float16)f.z, (_Float16)f.w};
        *(v4h*)&As[r * LPITCH + c4 * 4] = hh;
      }
    }

    // ---- stage B -> Bs[n][k] (128 x 32) ----
    if (BWT) {
      if (sizeof(TB) == 2) {
#pragma unroll
        for (int it = 0; it < 2; ++it) {
          int i = tid + it * 256;          // 512 chunks of 8 halves
          int n = i >> 2, c8 = i & 3;
          int gn = colBase + n;
          _Float16* dst = &Bs[n * LPITCH + c8 * 8];
          if (gn < p.N) {
            async_cp16((const void*)((const _Float16*)B + (long)gn * p.ldb + k0 + c8 * 8), dst);
          } else {
            *(v8h*)dst = (v8h)(_Float16)0.f;
          }
        }
      } else {
#pragma unroll
        for (int it = 0; it < 4; ++it) {
          int i = tid + it * 256;          // 1024 float4 chunks
          int n = i >> 3, c4 = i & 7;
          int gn = colBase + n;
          float4 f = make_float4(0.f, 0.f, 0.f, 0.f);
          if (gn < p.N) f = *(const float4*)((const float*)B + (long)gn * p.ldb + k0 + c4 * 4);
          v4h hh = {(_Float16)f.x, (_Float16)f.y, (_Float16)f.z, (_Float16)f.w};
          *(v4h*)&Bs[n * LPITCH + c4 * 4] = hh;
        }
      }
    } else {
      // B is (K,N): vector-load 8 consecutive n, transpose-store into [n][k]
#pragma unroll
      for (int it = 0; it < 2; ++it) {
        int i = tid + it * 256;            // 32 k-rows x 16 chunks
        int kk = i >> 4, j8 = (i & 15) * 8;
        int gn = colBase + j8;
        if (gn + 7 < p.N) {
          v8h vv = *(const v8h*)((const _Float16*)B + (long)(k0 + kk) * p.ldb + gn);
#pragma unroll
          for (int j = 0; j < 8; ++j) Bs[(j8 + j) * LPITCH + kk] = vv[j];
        } else {
#pragma unroll
          for (int j = 0; j < 8; ++j) {
            float v = (gn + j < p.N) ? (float)((const TB*)B)[(long)(k0 + kk) * p.ldb + gn + j] : 0.f;
            Bs[(j8 + j) * LPITCH + kk] = (_Float16)v;
          }
        }
      }
    }

    if (sizeof(TA) == 2 || (BWT && sizeof(TB) == 2)) async_wait();
    __syncthreads();

    v16h fa0 = frag_a(As, (waveM * 2 + 0) * 16, lo, hi);
    v16h fa1 = frag_a(As, (waveM * 2 + 1) * 16, lo, hi);
    v16h fb0 = frag_b(Bs, (waveN * 2 + 0) * 16, lo, hi);
    v16h fb1 = frag_b(Bs, (waveN * 2 + 1) * 16, lo, hi);
    acc[0][0] = __builtin_amdgcn_wmma_f32_16x16x32_f16(false, fa0, false, fb0, (short)0, acc[0][0], false, false);
    acc[0][1] = __builtin_amdgcn_wmma_f32_16x16x32_f16(false, fa0, false, fb1, (short)0, acc[0][1], false, false);
    acc[1][0] = __builtin_amdgcn_wmma_f32_16x16x32_f16(false, fa1, false, fb0, (short)0, acc[1][0], false, false);
    acc[1][1] = __builtin_amdgcn_wmma_f32_16x16x32_f16(false, fa1, false, fb1, (short)0, acc[1][1], false, false);
  }

  // epilogue: C/D layout -> lane lo = col, VGPR r = row r + 8*hi
#pragma unroll
  for (int tI = 0; tI < 2; ++tI) {
    const int gmBase = rowBase + (waveM * 2 + tI) * 16 + hi * 8;
#pragma unroll
    for (int tJ = 0; tJ < 2; ++tJ) {
      int gn = colBase + (waveN * 2 + tJ) * 16 + lo;
      if (gn >= p.N) continue;
#pragma unroll
      for (int r = 0; r < 8; ++r) {
        int gm = gmBase + r;
        if (gm >= p.M) continue;
        float v = acc[tI][tJ][r];
        if (p.bias) v += p.bias[gn];
        if (ACT == 2 || ACT == 3) v *= p.scale;
        if (ACT == 3) { if (gn > gm) v = -1.0e7f; }   // causal mask after scale
        if (p.res) v += p.res[p.bsRes * z + (long)gm * p.ldres + gn];
        if (ACT == 1) v = v > 0.f ? v : 0.01f * v;
        C[(long)gm * p.ldc + gn] = (TC)v;
      }
    }
  }
}

template<typename TA, typename TB, typename TC, bool BWT, int ACT>
static void launch_gemm(hipStream_t st,
    const void* A, long lda, long bsA,
    const void* B, long ldb, long bsB,
    const float* bias,
    const float* res, long ldres, long bsRes,
    void* C, long ldc, long bsC,
    int M, int N, int K, int nbatch, float scale)
{
  GemmP p{A, lda, bsA, B, ldb, bsB, bias, res, ldres, bsRes, C, ldc, bsC, M, N, K, scale};
  dim3 grid((N + 127) / 128, (M + 63) / 64, nbatch);
  gemm_kernel<TA, TB, TC, BWT, ACT><<<grid, dim3(256), 0, st>>>(p);
}

// ---------------------------------------------------------------------------
// Fused GRU step. grid.x = 16 column slabs of 64 hidden units.
// Computes gr = i_r+h_r, gz = i_z+h_z, i_n, h_n via WMMA over K=96 (x, W_ih)
// and K=1024 (h, W_hh), then the gate math, writing h_{t+1} into H[:,t+1,:].
// H layout: (B=64, S=512, 1024). X layout: (64, 512, 96).
// ---------------------------------------------------------------------------
__global__ void __launch_bounds__(256) gru_step_kernel(
    const float* __restrict__ X, float* __restrict__ H,
    const float* __restrict__ W_ih, const float* __restrict__ W_hh,
    const float* __restrict__ b_ih, const float* __restrict__ b_hh,
    int t)
{
  const int S = 512, HID = 1024, POSE = 96;
  __shared__ _Float16 As[64 * LPITCH];       // x/h tile (64 batches x 32 k)
  __shared__ _Float16 Bs[3 * 64 * LPITCH];   // 3 gate slabs of W (64 cols x 32 k)

  const int tid  = threadIdx.x;
  const int lane = tid & 31;
  const int wave = tid >> 5;
  const int lo   = lane & 15;
  const int hi   = lane >> 4;
  const int c0   = blockIdx.x * 64;
  const int ct   = wave & 3;    // 16-col tile within slab
  const int mh   = wave >> 2;   // 0/1 -> rows 0-31 / 32-63

  v8f aR[2] = {}, aZ[2] = {}, aIN[2] = {}, aHN[2] = {};

  // ---- phase A: x_t @ W_ih^T slabs (K = 96) -> gr, gz, i_n ----
  for (int k0 = 0; k0 < POSE; k0 += 32) {
    __syncthreads();
#pragma unroll
    for (int it = 0; it < 2; ++it) {
      int i = tid + it * 256;
      int r = i >> 3, c4 = i & 7;
      float4 f = *(const float4*)&X[(long)r * S * POSE + (long)t * POSE + k0 + c4 * 4];
      v4h hh = {(_Float16)f.x, (_Float16)f.y, (_Float16)f.z, (_Float16)f.w};
      *(v4h*)&As[r * LPITCH + c4 * 4] = hh;
    }
#pragma unroll
    for (int it = 0; it < 6; ++it) {
      int i = tid + it * 256;
      int a = i >> 9; int rem = i & 511; int n = rem >> 3, c4 = rem & 7;
      float4 f = *(const float4*)&W_ih[(long)(a * HID + c0 + n) * POSE + k0 + c4 * 4];
      v4h hh = {(_Float16)f.x, (_Float16)f.y, (_Float16)f.z, (_Float16)f.w};
      *(v4h*)&Bs[a * 64 * LPITCH + n * LPITCH + c4 * 4] = hh;
    }
    __syncthreads();
#pragma unroll
    for (int tI = 0; tI < 2; ++tI) {
      v16h fa = frag_a(As, mh * 32 + tI * 16, lo, hi);
      v16h f0 = frag_b(Bs + 0 * 64 * LPITCH, ct * 16, lo, hi);
      v16h f1 = frag_b(Bs + 1 * 64 * LPITCH, ct * 16, lo, hi);
      v16h f2 = frag_b(Bs + 2 * 64 * LPITCH, ct * 16, lo, hi);
      aR[tI]  = __builtin_amdgcn_wmma_f32_16x16x32_f16(false, fa, false, f0, (short)0, aR[tI],  false, false);
      aZ[tI]  = __builtin_amdgcn_wmma_f32_16x16x32_f16(false, fa, false, f1, (short)0, aZ[tI],  false, false);
      aIN[tI] = __builtin_amdgcn_wmma_f32_16x16x32_f16(false, fa, false, f2, (short)0, aIN[tI], false, false);
    }
  }

  // ---- phase B: h_t @ W_hh^T slabs (K = 1024) -> gr, gz, h_n ----
  for (int k0 = 0; k0 < HID; k0 += 32) {
    __syncthreads();
#pragma unroll
    for (int it = 0; it < 2; ++it) {
      int i = tid + it * 256;
      int r = i >> 3, c4 = i & 7;
      float4 f = *(const float4*)&H[(long)r * S * HID + (long)t * HID + k0 + c4 * 4];
      v4h hh = {(_Float16)f.x, (_Float16)f.y, (_Float16)f.z, (_Float16)f.w};
      *(v4h*)&As[r * LPITCH + c4 * 4] = hh;
    }
#pragma unroll
    for (int it = 0; it < 6; ++it) {
      int i = tid + it * 256;
      int a = i >> 9; int rem = i & 511; int n = rem >> 3, c4 = rem & 7;
      float4 f = *(const float4*)&W_hh[(long)(a * HID + c0 + n) * HID + k0 + c4 * 4];
      v4h hh = {(_Float16)f.x, (_Float16)f.y, (_Float16)f.z, (_Float16)f.w};
      *(v4h*)&Bs[a * 64 * LPITCH + n * LPITCH + c4 * 4] = hh;
    }
    __syncthreads();
#pragma unroll
    for (int tI = 0; tI < 2; ++tI) {
      v16h fa = frag_a(As, mh * 32 + tI * 16, lo, hi);
      v16h f0 = frag_b(Bs + 0 * 64 * LPITCH, ct * 16, lo, hi);
      v16h f1 = frag_b(Bs + 1 * 64 * LPITCH, ct * 16, lo, hi);
      v16h f2 = frag_b(Bs + 2 * 64 * LPITCH, ct * 16, lo, hi);
      aR[tI]  = __builtin_amdgcn_wmma_f32_16x16x32_f16(false, fa, false, f0, (short)0, aR[tI],  false, false);
      aZ[tI]  = __builtin_amdgcn_wmma_f32_16x16x32_f16(false, fa, false, f1, (short)0, aZ[tI],  false, false);
      aHN[tI] = __builtin_amdgcn_wmma_f32_16x16x32_f16(false, fa, false, f2, (short)0, aHN[tI], false, false);
    }
  }

  // ---- gate math + write h_{t+1} ----
  const int j = c0 + ct * 16 + lo;
  const float bir = b_ih[j]           + b_hh[j];
  const float biz = b_ih[HID + j]     + b_hh[HID + j];
  const float bin = b_ih[2 * HID + j];
  const float bhn = b_hh[2 * HID + j];
#pragma unroll
  for (int tI = 0; tI < 2; ++tI) {
#pragma unroll
    for (int r = 0; r < 8; ++r) {
      int m = mh * 32 + tI * 16 + hi * 8 + r;     // batch index
      float rg = 1.f / (1.f + __expf(-(aR[tI][r] + bir)));
      float zg = 1.f / (1.f + __expf(-(aZ[tI][r] + biz)));
      float ng = tanhf(aIN[tI][r] + bin + rg * (aHN[tI][r] + bhn));
      float hp = H[(long)m * S * HID + (long)t * HID + j];
      H[(long)m * S * HID + (long)(t + 1) * HID + j] = (1.f - zg) * ng + zg * hp;
    }
  }
}

// ---------------------------------------------------------------------------
// Row softmax over 512 columns, in place. grid.x = number of rows.
// ---------------------------------------------------------------------------
__global__ void __launch_bounds__(256) softmax512(float* __restrict__ SC) {
  const long row = blockIdx.x;
  float* p = SC + row * 512;
  const int tid = threadIdx.x;
  float v0 = p[tid], v1 = p[tid + 256];

  float m = fmaxf(v0, v1);
  for (int o = 16; o > 0; o >>= 1) m = fmaxf(m, __shfl_xor(m, o, 32));
  __shared__ float redm[8];
  if ((tid & 31) == 0) redm[tid >> 5] = m;
  __syncthreads();
  float mm = redm[0];
#pragma unroll
  for (int w = 1; w < 8; ++w) mm = fmaxf(mm, redm[w]);

  float e0 = __expf(v0 - mm), e1 = __expf(v1 - mm);
  float s = e0 + e1;
  for (int o = 16; o > 0; o >>= 1) s += __shfl_xor(s, o, 32);
  __shared__ float reds[8];
  if ((tid & 31) == 0) reds[tid >> 5] = s;
  __syncthreads();
  float ss = 0.f;
#pragma unroll
  for (int w = 0; w < 8; ++w) ss += reds[w];
  float inv = 1.f / ss;
  p[tid] = e0 * inv;
  p[tid + 256] = e1 * inv;
}

// ---------------------------------------------------------------------------
// Init kernels: H[:,0,:] = h ; X[:,0,:] = gt[:,0,:]
// ---------------------------------------------------------------------------
__global__ void init_h0(const float* __restrict__ h, float* __restrict__ H) {
  int i = blockIdx.x * 256 + threadIdx.x;
  if (i < 64 * 1024) {
    int b = i >> 10, d = i & 1023;
    H[(size_t)b * 512 * 1024 + d] = h[i];
  }
}
__global__ void init_x0(const float* __restrict__ gt, float* __restrict__ X) {
  int i = blockIdx.x * 256 + threadIdx.x;
  if (i < 64 * 96) {
    int b = i / 96, q = i % 96;
    X[(size_t)b * 512 * 96 + q] = gt[(size_t)b * 512 * 96 + q];
  }
}

// ---------------------------------------------------------------------------
// Host orchestration
// ---------------------------------------------------------------------------
extern "C" void kernel_launch(void* const* d_in, const int* in_sizes, int n_in,
                              void* d_out, int out_size, void* d_ws, size_t ws_size,
                              hipStream_t stream) {
  (void)in_sizes; (void)n_in; (void)out_size; (void)ws_size;
  const float* h_in = (const float*)d_in[0];
  const float* gt   = (const float*)d_in[1];
  const float* W_ih = (const float*)d_in[3];
  const float* W_hh = (const float*)d_in[4];
  const float* b_ih = (const float*)d_in[5];
  const float* b_hh = (const float*)d_in[6];
  const float* tp_W = (const float*)d_in[7];
  const float* tp_b = (const float*)d_in[8];
  const float* tQ_W = (const float*)d_in[9];
  const float* tQ_b = (const float*)d_in[10];
  const float* tK_W = (const float*)d_in[11];
  const float* tK_b = (const float*)d_in[12];
  const float* tV_W = (const float*)d_in[13];
  const float* tV_b = (const float*)d_in[14];
  const float* sQ_W = (const float*)d_in[15];
  const float* sQ_b = (const float*)d_in[16];
  const float* sK_W = (const float*)d_in[17];
  const float* sK_b = (const float*)d_in[18];
  const float* sV_W = (const float*)d_in[19];
  const float* sV_b = (const float*)d_in[20];
  const float* mlp_W = (const float*)d_in[21];
  const float* mlp_b = (const float*)d_in[22];
  const float* lin_W = (const float*)d_in[23];
  const float* lin_b = (const float*)d_in[24];
  float* out = (float*)d_out;

  const int S = 512, HID = 1024, POSE = 96, T = 511, NB = 64;
  const long HStride = (long)S * HID;          // per-batch stride in H
  const long SCb = (long)S * S;                // per-batch stride in scores
  const float scale = 0.03125f;                // 1024^-0.5

  // workspace carve-up (~524 MB)
  char* w = (char*)d_ws;
  float* H0 = (float*)w;        w += (size_t)NB * S * HID * 4;
  float* H1 = (float*)w;        w += (size_t)NB * S * HID * 4;
  float* X  = (float*)w;        w += (size_t)NB * S * POSE * 4;
  _Float16* Qh = (_Float16*)w;  w += (size_t)NB * S * HID * 2;
  _Float16* Kh = (_Float16*)w;  w += (size_t)NB * S * HID * 2;
  _Float16* Vh = (_Float16*)w;  w += (size_t)NB * S * HID * 2;
  float* SC = (float*)w;        w += (size_t)NB * S * S * 4;

  init_h0<<<dim3(256), dim3(256), 0, stream>>>(h_in, H0);
  init_x0<<<dim3(24), dim3(256), 0, stream>>>(gt, X);

  // ---- GRU scan (serial chain: 2 launches per step) ----
  for (int t = 0; t < T; ++t) {
    gru_step_kernel<<<dim3(16), dim3(256), 0, stream>>>(X, H0, W_ih, W_hh, b_ih, b_hh, t);
    // y = x_t + h_{t+1} @ tp_W^T + tp_b  ->  X[:,t+1,:]
    launch_gemm<float, float, float, true, 0>(stream,
        H0 + (size_t)(t + 1) * HID, HStride, 0,
        tp_W, HID, 0,
        tp_b,
        X + (size_t)t * POSE, (long)S * POSE, 0,
        X + (size_t)(t + 1) * POSE, (long)S * POSE, 0,
        NB, POSE, HID, 1, 1.f);
  }

  // ---- transformer layers ----
  float* cur = H0;
  float* alt = H1;
  for (int i = 0; i < 2; ++i) {
    const size_t wo  = (size_t)i * HID * HID;
    const size_t bo  = (size_t)i * HID;
    const size_t wo2 = (size_t)i * HID * POSE;

    // temporal Q/K/V projections (f32 in -> f16 out)
    launch_gemm<float, float, _Float16, true, 0>(stream, cur, HID, 0, tQ_W + wo, HID, 0,
        tQ_b + bo, nullptr, 0, 0, Qh, HID, 0, NB * S, HID, HID, 1, 1.f);
    launch_gemm<float, float, _Float16, true, 0>(stream, cur, HID, 0, tK_W + wo, HID, 0,
        tK_b + bo, nullptr, 0, 0, Kh, HID, 0, NB * S, HID, HID, 1, 1.f);
    launch_gemm<float, float, _Float16, true, 0>(stream, cur, HID, 0, tV_W + wo, HID, 0,
        tV_b + bo, nullptr, 0, 0, Vh, HID, 0, NB * S, HID, HID, 1, 1.f);
    // scores = scale * Q K^T with causal mask, per batch (async f16 staging)
    launch_gemm<_Float16, _Float16, float, true, 3>(stream, Qh, HID, HStride, Kh, HID, HStride,
        nullptr, nullptr, 0, 0, SC, S, SCb, S, S, HID, NB, scale);
    softmax512<<<dim3(NB * S), dim3(256), 0, stream>>>(SC);
    // H' = P @ V + H
    launch_gemm<float, _Float16, float, false, 0>(stream, SC, S, SCb, Vh, HID, HStride,
        nullptr, cur, HID, HStride, alt, HID, HStride, S, HID, S, NB, 1.f);
    { float* tmp = cur; cur = alt; alt = tmp; }

    // spatial attention: q from H, k/v from X (K = 96)
    launch_gemm<float, float, _Float16, true, 0>(stream, cur, HID, 0, sQ_W + wo, HID, 0,
        sQ_b + bo, nullptr, 0, 0, Qh, HID, 0, NB * S, HID, HID, 1, 1.f);
    launch_gemm<float, float, _Float16, true, 0>(stream, X, POSE, 0, sK_W + wo2, POSE, 0,
        sK_b + bo, nullptr, 0, 0, Kh, HID, 0, NB * S, HID, POSE, 1, 1.f);
    launch_gemm<float, float, _Float16, true, 0>(stream, X, POSE, 0, sV_W + wo2, POSE, 0,
        sV_b + bo, nullptr, 0, 0, Vh, HID, 0, NB * S, HID, POSE, 1, 1.f);
    launch_gemm<_Float16, _Float16, float, true, 2>(stream, Qh, HID, HStride, Kh, HID, HStride,
        nullptr, nullptr, 0, 0, SC, S, SCb, S, S, HID, NB, scale);
    softmax512<<<dim3(NB * S), dim3(256), 0, stream>>>(SC);
    launch_gemm<float, _Float16, float, false, 0>(stream, SC, S, SCb, Vh, HID, HStride,
        nullptr, cur, HID, HStride, alt, HID, HStride, S, HID, S, NB, 1.f);
    { float* tmp = cur; cur = alt; alt = tmp; }

    // MLP: H' = leaky_relu(H @ mlp_W^T + mlp_b)
    launch_gemm<float, float, float, true, 1>(stream, cur, HID, 0, mlp_W + wo, HID, 0,
        mlp_b + bo, nullptr, 0, 0, alt, HID, 0, NB * S, HID, HID, 1, 1.f);
    { float* tmp = cur; cur = alt; alt = tmp; }
  }

  // ---- final projection: out = H @ lin_W^T + lin_b ----
  launch_gemm<float, float, float, true, 0>(stream, cur, HID, 0, lin_W, HID, 0,
      lin_b, nullptr, 0, 0, out, POSE, 0, NB * S, POSE, HID, 1, 1.f);
}
